// FlowRegressor_55336358641937
// MI455X (gfx1250) — matrix operand order repaired and checked by reading, not compiled, not run on hardware
//
#include <hip/hip_runtime.h>
#include <hip/hip_bf16.h>

typedef _Float16 half_t;
typedef __attribute__((ext_vector_type(16))) _Float16 v16h;
typedef __attribute__((ext_vector_type(8)))  _Float16 v8h;
typedef __attribute__((ext_vector_type(8)))  float    v8f;
typedef __attribute__((ext_vector_type(4))) unsigned int u32x4;
typedef __attribute__((ext_vector_type(8))) int i32x8;
typedef __attribute__((ext_vector_type(4))) int i32x4;

// ---------------- problem constants ----------------
constexpr int BB   = 8;
constexpr int NN1  = 8192;
constexpr int NN4  = 2048;
constexpr int CC   = 128;
constexpr int KNB  = 32;     // neighbors
constexpr int CIN1 = 131;    // 3 + C
constexpr int KPAD = 160;    // 131 padded to multiple of 32

// ---------------- fragment load helpers (CDNA5 ISA 7.12.2 layouts) ----------------
__device__ __forceinline__ v16h cat16(v8h lo, v8h hi) {
  return __builtin_shufflevector(lo, hi, 0, 1, 2, 3, 4, 5, 6, 7,
                                         8, 9, 10, 11, 12, 13, 14, 15);
}
// A 16x32 f16: lane's elems 0..7 = K k0..k0+7, elems 8..15 = K k0+16..k0+23
__device__ __forceinline__ v16h load_fragA(const half_t* p) {
  return cat16(*(const v8h*)p, *(const v8h*)(p + 16));
}
// B 32x16 f16: lane's elems 0..7 = K k0..k0+7, elems 8..15 = K k0+16..k0+23
__device__ __forceinline__ v16h load_fragB(const half_t* p) {
  return cat16(*(const v8h*)p, *(const v8h*)(p + 16));
}

// ---------------- TDM: DMA a contiguous f16 panel (1-D tensor) into LDS ----------
// D# per CDNA5 ISA ch.8: group0 {count=1, lds_addr, global_addr, type=2},
// group1 {data_size=2B, tensor_dim0=n, tensor_dim1=1, tile_dim0=n, tile_dim1=1,
//         tensor_dim0_stride=n}.  Issued by one lane; TDM ignores EXEC.
__device__ __forceinline__ void tdm_load_panel(const half_t* gptr,
                                               unsigned lds_off, int nhalves) {
#if __has_builtin(__builtin_amdgcn_tensor_load_to_lds)
  unsigned long long ga = (unsigned long long)(size_t)gptr;
  u32x4 g0;
  g0[0] = 1u;                                                  // count = 1 (valid)
  g0[1] = lds_off;                                             // lds_addr
  g0[2] = (unsigned)(ga & 0xffffffffu);                        // global_addr[31:0]
  g0[3] = (unsigned)((ga >> 32) & 0x1ffffffu) | (2u << 30);    // addr[56:32] | type=2
  i32x8 g1;
  g1[0] = 0x10000;                               // data_size=1 -> 2 bytes
  g1[1] = (int)(((unsigned)nhalves & 0xffffu) << 16);          // tensor_dim0[15:0]
  g1[2] = (int)((((unsigned)nhalves >> 16) & 0xffffu) | (1u << 16)); // dim0 hi | tensor_dim1=1
  g1[3] = (int)(((unsigned)nhalves & 0xffffu) << 16);          // tile_dim0
  g1[4] = 1;                                                   // tile_dim1=1
  g1[5] = nhalves;                                             // tensor_dim0_stride lo
  g1[6] = 0;
  g1[7] = 0;
  i32x4 z4 = {0, 0, 0, 0};
#if defined(__clang_major__) && (__clang_major__ >= 23)
  i32x8 z8 = {0, 0, 0, 0, 0, 0, 0, 0};
  __builtin_amdgcn_tensor_load_to_lds(g0, g1, z4, z4, z8, 0);
#else
  __builtin_amdgcn_tensor_load_to_lds(g0, g1, z4, z4, 0);
#endif
#endif
}

// ---------------- layout / transpose preps ----------------
__global__ void transpose_f32_kernel(const float* __restrict__ in, float* __restrict__ out,
                                     int Bc, int Cc, int Nc) {
  int i = blockIdx.x * 256 + threadIdx.x;
  if (i >= Bc * Cc * Nc) return;
  int n = i % Nc; int t = i / Nc; int c = t % Cc; int b = t / Cc;
  out[((size_t)b * Nc + n) * Cc + c] = in[i];
}

__global__ void transpose_cast_kernel(const float* __restrict__ in, half_t* __restrict__ out,
                                      int Bc, int Cc, int Nc) {
  int i = blockIdx.x * 256 + threadIdx.x;
  if (i >= Bc * Cc * Nc) return;
  int n = i % Nc; int t = i / Nc; int c = t % Cc; int b = t / Cc;
  out[((size_t)b * Nc + n) * Cc + c] = (half_t)in[i];
}

// W: [Cout=CC, Cin] row-major -> Wp: [CC, Kp] f16  (K-major per output column)
// mode 0 (plain):  Wp[c][k] = W[c][k]                      (k < Cin else 0)
// mode 1 (gather): A cols reordered to [feat(128), xyz(3), pad]:
//                  k<128 -> W[c][k+3]; 128..130 -> W[c][k-128]; else 0
__global__ void prep_w_kernel(const float* __restrict__ W, half_t* __restrict__ Wp,
                              int Cin, int Kp, int mode) {
  int i = blockIdx.x * 256 + threadIdx.x;
  if (i >= CC * Kp) return;
  int k = i % Kp;
  int c = i / Kp;
  float v;
  if (mode == 0) {
    v = (k < Cin) ? W[c * Cin + k] : 0.f;
  } else {
    if (k < 128)        v = W[c * Cin + (k + 3)];
    else if (k < 131)   v = W[c * Cin + (k - 128)];
    else                v = 0.f;
  }
  Wp[i] = (half_t)v;
}

// ---------------- kNN: per-thread streaming 32-NN over 2048 points ----------------
__global__ void knn_kernel(const float* __restrict__ xyz, int* __restrict__ knn) {
  int i = blockIdx.x * 256 + threadIdx.x;   // b*NN4 + n
  if (i >= BB * NN4) return;
  int b = i / NN4;
  const float* base = xyz + (size_t)b * NN4 * 3;
  float qx = xyz[(size_t)i * 3 + 0];
  float qy = xyz[(size_t)i * 3 + 1];
  float qz = xyz[(size_t)i * 3 + 2];
  float bd[KNB]; int bi[KNB];
  #pragma unroll
  for (int t = 0; t < KNB; ++t) { bd[t] = 3.4e38f; bi[t] = 0; }
  float wmax = 3.4e38f; int wpos = 0;
  for (int j = 0; j < NN4; ++j) {
    float dx = base[j * 3 + 0] - qx;
    float dy = base[j * 3 + 1] - qy;
    float dz = base[j * 3 + 2] - qz;
    float d = dx * dx + dy * dy + dz * dz;
    if (d < wmax) {
      bd[wpos] = d; bi[wpos] = j;
      wmax = -1.f;
      #pragma unroll
      for (int t = 0; t < KNB; ++t) if (bd[t] > wmax) { wmax = bd[t]; wpos = t; }
    }
  }
  #pragma unroll
  for (int t = 0; t < KNB; ++t) knn[(size_t)i * KNB + t] = bi[t];
}

// ---------------- GEMM 1: gathered concat [feat(128), dxyz(3)] x Wp ----------------
__global__ __launch_bounds__(256) void gemm_gather_kernel(
    const float*  __restrict__ xyz,     // [B,N4,3]
    const half_t* __restrict__ feats,   // [B,N4,C]
    const int*    __restrict__ knn,     // [B,N4,K]
    const half_t* __restrict__ Wp,      // [CC, KPAD]  (K-major)
    const float*  __restrict__ bias,    // [C]
    float* __restrict__ y,              // [B*N4*K, C]
    float* __restrict__ ssum, float* __restrict__ ssq)
{
  __shared__ alignas(16) half_t sA[64 * KPAD];
  __shared__ alignas(16) half_t sW[CC * KPAD];
  __shared__ float sBias[CC];
  const int tid = threadIdx.x;
  const int m0  = blockIdx.x * 64;

  // stage W: TDM async DMA (one descriptor), fallback = vector copy
#if __has_builtin(__builtin_amdgcn_tensor_load_to_lds)
  if (tid == 0)
    tdm_load_panel(Wp, (unsigned)(size_t)&sW[0], CC * KPAD);
  if (tid < 32)
    __builtin_amdgcn_s_wait_tensorcnt((short)0);
#else
  for (int i = tid * 8; i < CC * KPAD; i += 256 * 8)
    *(v8h*)(sW + i) = *(const v8h*)(Wp + i);
#endif
  if (tid < CC) sBias[tid] = bias[tid];

  // stage A: 64 rows x [feat(128) | dxyz(3) | zero-pad(29)]
  {
    const int r   = tid >> 2;   // 0..63
    const int seg = tid & 3;
    const int gm  = m0 + r;
    const int b   = gm / (NN4 * KNB);
    const int rem = gm - b * (NN4 * KNB);
    const int n   = rem / KNB;
    const int j   = knn[gm];
    const half_t* fj = feats + ((size_t)b * NN4 + j) * CC;
    const v8h* src = (const v8h*)(fj + seg * 32);
    v8h* dst = (v8h*)(sA + r * KPAD + seg * 32);
    dst[0] = src[0]; dst[1] = src[1]; dst[2] = src[2]; dst[3] = src[3];
    if (seg == 0) {
      const float* pq = xyz + ((size_t)b * NN4 + n) * 3;
      const float* pj = xyz + ((size_t)b * NN4 + j) * 3;
      half_t* tail = sA + r * KPAD + CC;
      tail[0] = (half_t)(pj[0] - pq[0]);
      tail[1] = (half_t)(pj[1] - pq[1]);
      tail[2] = (half_t)(pj[2] - pq[2]);
      #pragma unroll
      for (int t = 3; t < KPAD - CC; ++t) tail[t] = (half_t)0.f;
    }
  }
  __syncthreads();

  const int wave  = tid >> 5;
  const int lane  = tid & 31;
  const int msub  = wave & 3;
  const int ntb   = wave >> 2;
  const int lhalf = lane >> 4;
  const int l15   = lane & 15;
  const int b_idx = m0 / (NN4 * KNB);
  constexpr int KT = KPAD / 32;

  // A fragments shared across all n-tiles of this wave
  const half_t* arow = sA + (msub * 16 + l15) * KPAD + lhalf * 8;
  v16h afr[KT];
  #pragma unroll
  for (int kt = 0; kt < KT; ++kt) afr[kt] = load_fragA(arow + kt * 32);

  for (int nt = ntb; nt < 8; nt += 2) {
    const half_t* bcol = sW + (nt * 16 + l15) * KPAD + lhalf * 8;
    // batch all B fragments, then chain WMMAs back-to-back
    v16h bfr[KT];
    #pragma unroll
    for (int kt = 0; kt < KT; ++kt) bfr[kt] = load_fragB(bcol + kt * 32);
    v8f acc = {};
    #pragma unroll
    for (int kt = 0; kt < KT; ++kt)
      acc = __builtin_amdgcn_wmma_f32_16x16x32_f16(false, afr[kt], false, bfr[kt],
                                                   (short)0, acc, false, false);
    const int c = nt * 16 + l15;
    const float bv = sBias[c];
    float s = 0.f, s2 = 0.f;
    #pragma unroll
    for (int r = 0; r < 8; ++r) {
      const int ml = r + lhalf * 8;
      const float val = acc[r] + bv;
      y[(size_t)(m0 + msub * 16 + ml) * CC + c] = val;
      s += val; s2 += val * val;
    }
    s  += __shfl_xor(s, 16, 32);
    s2 += __shfl_xor(s2, 16, 32);
    if (lhalf == 0) {
      atomicAdd(&ssum[b_idx * CC + c], s);
      atomicAdd(&ssq [b_idx * CC + c], s2);
    }
  }
}

// ---------------- GEMM 2: plain f16 [Mtot,128] x Wp(128x128) ----------------
template <bool WITH_STATS>
__global__ __launch_bounds__(256) void gemm_plain_kernel(
    const half_t* __restrict__ A,      // [Mtot, C]
    const half_t* __restrict__ Wp,     // [CC, CC]  (K-major)
    const float*  __restrict__ bias,   // [C]
    float*  __restrict__ y,            // WITH_STATS: f32 out
    half_t* __restrict__ yh,           // !WITH_STATS: relu f16 out
    float* __restrict__ ssum, float* __restrict__ ssq,
    int rows_per_b)
{
  __shared__ alignas(16) half_t sA[64 * CC];
  __shared__ alignas(16) half_t sW[CC * CC];
  __shared__ float sBias[CC];
  const int tid = threadIdx.x;
  const int m0  = blockIdx.x * 64;

#if __has_builtin(__builtin_amdgcn_tensor_load_to_lds)
  if (tid == 0)
    tdm_load_panel(Wp, (unsigned)(size_t)&sW[0], CC * CC);
  if (tid < 32)
    __builtin_amdgcn_s_wait_tensorcnt((short)0);
#else
  for (int i = tid * 8; i < CC * CC; i += 256 * 8)
    *(v8h*)(sW + i) = *(const v8h*)(Wp + i);
#endif
  if (tid < CC) sBias[tid] = bias[tid];
  {
    const int r   = tid >> 2;
    const int seg = tid & 3;
    const v8h* src = (const v8h*)(A + (size_t)(m0 + r) * CC + seg * 32);
    v8h* dst = (v8h*)(sA + r * CC + seg * 32);
    dst[0] = src[0]; dst[1] = src[1]; dst[2] = src[2]; dst[3] = src[3];
  }
  __syncthreads();

  const int wave  = tid >> 5;
  const int lane  = tid & 31;
  const int msub  = wave & 3;
  const int ntb   = wave >> 2;
  const int lhalf = lane >> 4;
  const int l15   = lane & 15;
  const int b_idx = m0 / rows_per_b;
  constexpr int KT = CC / 32;

  const half_t* arow = sA + (msub * 16 + l15) * CC + lhalf * 8;
  v16h afr[KT];
  #pragma unroll
  for (int kt = 0; kt < KT; ++kt) afr[kt] = load_fragA(arow + kt * 32);

  for (int nt = ntb; nt < 8; nt += 2) {
    const half_t* bcol = sW + (nt * 16 + l15) * CC + lhalf * 8;
    v16h bfr[KT];
    #pragma unroll
    for (int kt = 0; kt < KT; ++kt) bfr[kt] = load_fragB(bcol + kt * 32);
    v8f acc = {};
    #pragma unroll
    for (int kt = 0; kt < KT; ++kt)
      acc = __builtin_amdgcn_wmma_f32_16x16x32_f16(false, afr[kt], false, bfr[kt],
                                                   (short)0, acc, false, false);
    const int c = nt * 16 + l15;
    const float bv = sBias[c];
    float s = 0.f, s2 = 0.f;
    #pragma unroll
    for (int r = 0; r < 8; ++r) {
      const int ml = r + lhalf * 8;
      const float val = acc[r] + bv;
      const size_t row = (size_t)(m0 + msub * 16 + ml);
      if (WITH_STATS) {
        y[row * CC + c] = val;
        s += val; s2 += val * val;
      } else {
        yh[row * CC + c] = (half_t)(val > 0.f ? val : 0.f);
      }
    }
    if (WITH_STATS) {
      s  += __shfl_xor(s, 16, 32);
      s2 += __shfl_xor(s2, 16, 32);
      if (lhalf == 0) {
        atomicAdd(&ssum[b_idx * CC + c], s);
        atomicAdd(&ssq [b_idx * CC + c], s2);
      }
    }
  }
}

// ---------------- InstanceNorm apply (+relu), full tensor, f16 out ----------------
__global__ void norm_relu_kernel(const float* __restrict__ y,
                                 const float* __restrict__ ssum,
                                 const float* __restrict__ ssq,
                                 half_t* __restrict__ out, int total) {
  int i = blockIdx.x * 256 + threadIdx.x;
  if (i >= total) return;
  int c = i & (CC - 1);
  int b = i / (NN4 * KNB * CC);
  const float cnt = (float)(NN4 * KNB);
  float m = ssum[b * CC + c] / cnt;
  float v = ssq [b * CC + c] / cnt - m * m;
  float inv = rsqrtf(v + 1e-5f);
  float val = (y[i] - m) * inv;
  out[i] = (half_t)(val > 0.f ? val : 0.f);
}

// ---------------- InstanceNorm apply (+relu) + max-pool over K ----------------
__global__ void norm_relu_max_kernel(const float* __restrict__ y,
                                     const float* __restrict__ ssum,
                                     const float* __restrict__ ssq,
                                     half_t* __restrict__ x) {
  int i = blockIdx.x * 256 + threadIdx.x;   // (b*N4+n)*C + c
  if (i >= BB * NN4 * CC) return;
  int c  = i & (CC - 1);
  int bn = i >> 7;
  int b  = bn / NN4;
  const float cnt = (float)(NN4 * KNB);
  float m = ssum[b * CC + c] / cnt;
  float v = ssq [b * CC + c] / cnt - m * m;
  float inv = rsqrtf(v + 1e-5f);
  const float* p = y + (size_t)bn * KNB * CC + c;
  float mx = 0.f;   // relu floor
  for (int k = 0; k < KNB; ++k) {
    float val = (p[(size_t)k * CC] - m) * inv;
    if (val > mx) mx = val;
  }
  x[i] = (half_t)mx;
}

// ---------------- FC2 (128 -> 3) + transposed low-res output ----------------
__global__ void fc2_kernel(const half_t* __restrict__ xin,   // [B*N4, C]
                           const float* __restrict__ W2,     // [3, C]
                           const float* __restrict__ b2,     // [3]
                           float* __restrict__ flowlr,       // [B,N4,3]
                           float* __restrict__ out_lr) {     // d_out: [B,3,N4]
  int i = blockIdx.x * 256 + threadIdx.x;    // b*N4 + n
  if (i >= BB * NN4) return;
  int b = i / NN4, n = i - b * NN4;
  const half_t* p = xin + (size_t)i * CC;
  for (int d = 0; d < 3; ++d) {
    float s = b2[d];
    for (int c = 0; c < CC; ++c) s += (float)p[c] * W2[d * CC + c];
    flowlr[(size_t)i * 3 + d] = s;
    out_lr[((size_t)b * 3 + d) * NN4 + n] = s;
  }
}

// ---------------- feature propagation: streaming 3-NN interp ----------------
__global__ void fp_kernel(const float* __restrict__ xyz1,    // [B,N1,3]
                          const float* __restrict__ xyz4,    // [B,N4,3]
                          const float* __restrict__ flowlr,  // [B,N4,3]
                          float* __restrict__ out_flow) {    // d_out: [B,3,N1]
  int i = blockIdx.x * 256 + threadIdx.x;    // b*N1 + n
  if (i >= BB * NN1) return;
  int b = i / NN1, n = i - b * NN1;
  const float qx = xyz1[(size_t)i * 3 + 0];
  const float qy = xyz1[(size_t)i * 3 + 1];
  const float qz = xyz1[(size_t)i * 3 + 2];
  const float* base = xyz4 + (size_t)b * NN4 * 3;
  float d0 = 3.4e38f, d1 = 3.4e38f, d2 = 3.4e38f;
  int i0 = 0, i1 = 0, i2 = 0;
  for (int j = 0; j < NN4; ++j) {
    float dx = base[j * 3 + 0] - qx;
    float dy = base[j * 3 + 1] - qy;
    float dz = base[j * 3 + 2] - qz;
    float d = dx * dx + dy * dy + dz * dz;
    if (d < d0)      { d2 = d1; i2 = i1; d1 = d0; i1 = i0; d0 = d; i0 = j; }
    else if (d < d1) { d2 = d1; i2 = i1; d1 = d;  i1 = j; }
    else if (d < d2) { d2 = d;  i2 = j; }
  }
  float w0 = 1.f / (d0 + 1e-8f), w1 = 1.f / (d1 + 1e-8f), w2 = 1.f / (d2 + 1e-8f);
  float wsum = w0 + w1 + w2;
  w0 /= wsum; w1 /= wsum; w2 /= wsum;
  const float* f0 = flowlr + ((size_t)b * NN4 + i0) * 3;
  const float* f1 = flowlr + ((size_t)b * NN4 + i1) * 3;
  const float* f2 = flowlr + ((size_t)b * NN4 + i2) * 3;
  for (int d = 0; d < 3; ++d) {
    out_flow[((size_t)b * 3 + d) * NN1 + n] = w0 * f0[d] + w1 * f1[d] + w2 * f2[d];
  }
}

// ---------------- host orchestration ----------------
extern "C" void kernel_launch(void* const* d_in, const int* in_sizes, int n_in,
                              void* d_out, int out_size, void* d_ws, size_t ws_size,
                              hipStream_t stream) {
  const float* xyz_s1 = (const float*)d_in[0];
  const float* xyz_s4 = (const float*)d_in[1];
  const float* feats  = (const float*)d_in[2];
  const float* sa1_W1 = (const float*)d_in[3];
  const float* sa1_b1 = (const float*)d_in[4];
  const float* sa1_W2 = (const float*)d_in[5];
  const float* sa1_b2 = (const float*)d_in[6];
  const float* sa2_W1 = (const float*)d_in[7];
  const float* sa2_b1 = (const float*)d_in[8];
  const float* sa2_W2 = (const float*)d_in[9];
  const float* sa2_b2 = (const float*)d_in[10];
  const float* fc_W1  = (const float*)d_in[11];
  const float* fc_b1  = (const float*)d_in[12];
  const float* fc_W2  = (const float*)d_in[13];
  const float* fc_b2  = (const float*)d_in[14];

  char* base = (char*)d_ws;
  size_t off = 0;
  auto alloc = [&](size_t bytes) -> char* {
    char* p = base + off;
    off = (off + bytes + 255) & ~(size_t)255;
    return p;
  };

  float*  xyz1   = (float*) alloc((size_t)BB * NN1 * 3 * 4);
  float*  xyz4   = (float*) alloc((size_t)BB * NN4 * 3 * 4);
  half_t* featTh = (half_t*)alloc((size_t)BB * NN4 * CC * 2);
  int*    knn    = (int*)   alloc((size_t)BB * NN4 * KNB * 4);
  half_t* w_g1   = (half_t*)alloc((size_t)CC * KPAD * 2);
  half_t* w_p1   = (half_t*)alloc((size_t)CC * CC * 2);
  half_t* w_g2   = (half_t*)alloc((size_t)CC * KPAD * 2);
  half_t* w_p2   = (half_t*)alloc((size_t)CC * CC * 2);
  half_t* w_fc1  = (half_t*)alloc((size_t)CC * CC * 2);
  float*  ybuf   = (float*) alloc((size_t)BB * NN4 * KNB * CC * 4);
  half_t* hn     = (half_t*)alloc((size_t)BB * NN4 * KNB * CC * 2);
  half_t* x1h    = (half_t*)alloc((size_t)BB * NN4 * CC * 2);
  half_t* x2h    = (half_t*)alloc((size_t)BB * NN4 * CC * 2);
  half_t* fc1h   = (half_t*)alloc((size_t)BB * NN4 * CC * 2);
  float*  flowlr = (float*) alloc((size_t)BB * NN4 * 3 * 4);
  float*  ssum   = (float*) alloc((size_t)BB * CC * 4);
  float*  ssq    = (float*) alloc((size_t)BB * CC * 4);
  (void)ws_size; (void)in_sizes; (void)n_in; (void)out_size;

  float* out_flow = (float*)d_out;                        // [B,3,N1]
  float* out_lr   = (float*)d_out + (size_t)BB * 3 * NN1; // [B,3,N4]

  const int Mtot = BB * NN4 * KNB;       // 524288 rows
  const int Mfc  = BB * NN4;             // 16384 rows
  const int totY = Mtot * CC;            // 67.1M elements

  // preps
  transpose_f32_kernel<<<(BB * 3 * NN1 + 255) / 256, 256, 0, stream>>>(xyz_s1, xyz1, BB, 3, NN1);
  transpose_f32_kernel<<<(BB * 3 * NN4 + 255) / 256, 256, 0, stream>>>(xyz_s4, xyz4, BB, 3, NN4);
  transpose_cast_kernel<<<(BB * CC * NN4 + 255) / 256, 256, 0, stream>>>(feats, featTh, BB, CC, NN4);
  knn_kernel<<<(BB * NN4 + 255) / 256, 256, 0, stream>>>(xyz4, knn);
  prep_w_kernel<<<(CC * KPAD + 255) / 256, 256, 0, stream>>>(sa1_W1, w_g1, CIN1, KPAD, 1);
  prep_w_kernel<<<(CC * CC + 255) / 256, 256, 0, stream>>>(sa1_W2, w_p1, CC, CC, 0);
  prep_w_kernel<<<(CC * KPAD + 255) / 256, 256, 0, stream>>>(sa2_W1, w_g2, CIN1, KPAD, 1);
  prep_w_kernel<<<(CC * CC + 255) / 256, 256, 0, stream>>>(sa2_W2, w_p2, CC, CC, 0);
  prep_w_kernel<<<(CC * CC + 255) / 256, 256, 0, stream>>>(fc_W1, w_fc1, CC, CC, 0);

  // ---- SA1 ----
  hipMemsetAsync(ssum, 0, (size_t)BB * CC * 4, stream);
  hipMemsetAsync(ssq,  0, (size_t)BB * CC * 4, stream);
  gemm_gather_kernel<<<Mtot / 64, 256, 0, stream>>>(xyz4, featTh, knn, w_g1, sa1_b1, ybuf, ssum, ssq);
  norm_relu_kernel<<<(totY + 255) / 256, 256, 0, stream>>>(ybuf, ssum, ssq, hn, totY);
  hipMemsetAsync(ssum, 0, (size_t)BB * CC * 4, stream);
  hipMemsetAsync(ssq,  0, (size_t)BB * CC * 4, stream);
  gemm_plain_kernel<true><<<Mtot / 64, 256, 0, stream>>>(hn, w_p1, sa1_b2, ybuf, nullptr, ssum, ssq, NN4 * KNB);
  norm_relu_max_kernel<<<(BB * NN4 * CC + 255) / 256, 256, 0, stream>>>(ybuf, ssum, ssq, x1h);

  // ---- SA2 ----
  hipMemsetAsync(ssum, 0, (size_t)BB * CC * 4, stream);
  hipMemsetAsync(ssq,  0, (size_t)BB * CC * 4, stream);
  gemm_gather_kernel<<<Mtot / 64, 256, 0, stream>>>(xyz4, x1h, knn, w_g2, sa2_b1, ybuf, ssum, ssq);
  norm_relu_kernel<<<(totY + 255) / 256, 256, 0, stream>>>(ybuf, ssum, ssq, hn, totY);
  hipMemsetAsync(ssum, 0, (size_t)BB * CC * 4, stream);
  hipMemsetAsync(ssq,  0, (size_t)BB * CC * 4, stream);
  gemm_plain_kernel<true><<<Mtot / 64, 256, 0, stream>>>(hn, w_p2, sa2_b2, ybuf, nullptr, ssum, ssq, NN4 * KNB);
  norm_relu_max_kernel<<<(BB * NN4 * CC + 255) / 256, 256, 0, stream>>>(ybuf, ssum, ssq, x2h);

  // ---- FC head ----
  gemm_plain_kernel<false><<<Mfc / 64, 256, 0, stream>>>(x2h, w_fc1, fc_b1, nullptr, fc1h, nullptr, nullptr, Mfc);
  fc2_kernel<<<(BB * NN4 + 255) / 256, 256, 0, stream>>>(fc1h, fc_W2, fc_b2, flowlr, out_lr);

  // ---- feature propagation ----
  fp_kernel<<<(BB * NN1 + 255) / 256, 256, 0, stream>>>(xyz1, xyz4, flowlr, out_flow);
}
